// MultiheadCrossAttention_14869176779073
// MI455X (gfx1250) — matrix operand order repaired and checked by reading, not compiled, not run on hardware
//
#include <hip/hip_runtime.h>
#include <hip/hip_bf16.h>

typedef __attribute__((ext_vector_type(16))) __bf16 v16bf;
typedef __attribute__((ext_vector_type(8)))  __bf16 v8bf;
typedef __attribute__((ext_vector_type(8)))  float  v8f;
typedef __attribute__((ext_vector_type(4)))  float  v4f;

constexpr int kB    = 4;
constexpr int kS    = 2048;
constexpr int kDin  = 512;
constexpr int kH    = 8;
constexpr int kDh   = 64;
constexpr int kHid  = 512;     // kH * kDh
constexpr int kOut2 = 2048;
constexpr int kM    = kB * kS; // 8192 rows

// ---------------------------------------------------------------------------
// helpers
// ---------------------------------------------------------------------------
__device__ inline v16bf cat8(v8bf lo, v8bf hi) {
  return __builtin_shufflevector(lo, hi, 0,1,2,3,4,5,6,7,8,9,10,11,12,13,14,15);
}

__device__ inline v8f wmma_bf16(v16bf a, v16bf b, v8f c) {
  // (neg_a, A, neg_b, B, c_mod, C, reuse_a, reuse_b)
  return __builtin_amdgcn_wmma_f32_16x16x32_bf16(false, a, false, b, (short)0, c, false, false);
}

// async 16B/lane global -> LDS copy (GLOBAL_LOAD_ASYNC_TO_LDS_B128, ASYNCcnt)
__device__ inline void async_g2l_b128(unsigned lds_off, const void* gaddr) {
  asm volatile("global_load_async_to_lds_b128 %0, %1, off"
               :: "v"(lds_off), "v"((unsigned long long)(size_t)gaddr)
               : "memory");
}
__device__ inline void wait_asynccnt0() {
  asm volatile("s_wait_asynccnt 0x0" ::: "memory");
}

// A-fragment 16x32 bf16 from fp32 row-major (lane L<16: row L, K=kb..kb+7 and
// 16+kb..16+kb+7 with kb = 8*(lane>=16))  -- two float4 loads per half
__device__ inline v16bf load_a_f32v(const float* __restrict__ A, int lda,
                                    int row0, int k0, int lane) {
  const int r  = lane & 15;
  const int kb = (lane >> 4) << 3;
  const float* p = A + (size_t)(row0 + r) * lda + k0 + kb;
  v4f x0 = *(const v4f*)(p);
  v4f x1 = *(const v4f*)(p + 4);
  v4f x2 = *(const v4f*)(p + 16);
  v4f x3 = *(const v4f*)(p + 20);
  v16bf a;
#pragma unroll
  for (int i = 0; i < 4; ++i) {
    a[i]      = (__bf16)x0[i];
    a[4 + i]  = (__bf16)x1[i];
    a[8 + i]  = (__bf16)x2[i];
    a[12 + i] = (__bf16)x3[i];
  }
  return a;
}

// A-fragment 16x32 from bf16 row-major source (two b128 loads)
__device__ inline v16bf load_a_bf16v(const __bf16* __restrict__ A, int lda,
                                     int row0, int k0, int lane) {
  const int r  = lane & 15;
  const int kb = (lane >> 4) << 3;
  const __bf16* p = A + (size_t)(row0 + r) * lda + k0 + kb;
  v8bf lo = *(const v8bf*)(p);
  v8bf hi = *(const v8bf*)(p + 16);
  return cat8(lo, hi);
}

// Cooperative stage of a 32x16 fp32 weight tile into LDS as bf16, stored
// col-major Bt[c][k] (c=0..15, k=0..31) so fragment reads are contiguous.
__device__ inline void stage_b_tile(const float* __restrict__ W, int ldw,
                                    int k0, int col0, __bf16* __restrict__ Bt,
                                    int tid) {
  const int c = tid & 15;
  const int k = tid >> 4;   // 0..15
  Bt[c * 32 + k]      = (__bf16)W[(size_t)(k0 + k) * ldw + col0 + c];
  Bt[c * 32 + k + 16] = (__bf16)W[(size_t)(k0 + k + 16) * ldw + col0 + c];
}

// B-fragment 32x16 from LDS tile Bt[c][k] (two ds b128 loads per lane)
__device__ inline v16bf read_b_frag(const __bf16* __restrict__ Bt, int lane) {
  const int c  = lane & 15;
  const int kb = (lane >> 4) << 3;
  const __bf16* p = Bt + c * 32 + kb;
  return cat8(*(const v8bf*)p, *(const v8bf*)(p + 16));
}

// ---------------------------------------------------------------------------
// Kernel 1: QKV projections. Block = 8 waves = 128 rows x 16 cols, LDS-shared
// double-buffered B tile. Output bf16 in [B,H,S,DH].
// ---------------------------------------------------------------------------
__global__ void __launch_bounds__(256)
qkv_kernel(const float* __restrict__ in1, const float* __restrict__ in2,
           const float* __restrict__ Wq, const float* __restrict__ bq,
           const float* __restrict__ Wk, const float* __restrict__ bk,
           const float* __restrict__ Wv, const float* __restrict__ bv,
           __bf16* __restrict__ Qb, __bf16* __restrict__ Kb, __bf16* __restrict__ Vb) {
  __shared__ __bf16 Bt[2][16 * 32];
  const int tid = threadIdx.x, lane = tid & 31, wisl = tid >> 5;
  constexpr int NT  = kHid / 16;            // 32 col tiles
  constexpr int BPM = (kM / 128) * NT;      // 2048 blocks per matrix
  const int mat  = blockIdx.x / BPM;        // 0=Q 1=K 2=V
  const int rem  = blockIdx.x % BPM;
  const int row0 = (rem / NT) * 128 + wisl * 16;
  const int col0 = (rem % NT) * 16;

  const float* X    = (mat == 0) ? in1 : in2;
  const float* W    = (mat == 0) ? Wq : (mat == 1) ? Wk : Wv;
  const float* bias = (mat == 0) ? bq : (mat == 1) ? bk : bv;
  __bf16*      O    = (mat == 0) ? Qb : (mat == 1) ? Kb : Vb;

  stage_b_tile(W, kHid, 0, col0, Bt[0], tid);
  __syncthreads();

  v8f acc = {};
  constexpr int NK = kDin / 32;             // 16
  for (int kc = 0; kc < NK; ++kc) {
    if (kc + 1 < NK) stage_b_tile(W, kHid, (kc + 1) * 32, col0, Bt[(kc + 1) & 1], tid);
    v16bf a = load_a_f32v(X, kDin, row0, kc * 32, lane);
    v16bf b = read_b_frag(Bt[kc & 1], lane);
    acc = wmma_bf16(a, b, acc);
    __syncthreads();
  }

  const int n  = col0 + (lane & 15);
  const int mo = (lane >> 4) << 3;
  const float bn = bias[n];
  const int h = n >> 6, d = n & (kDh - 1);
#pragma unroll
  for (int r = 0; r < 8; ++r) {
    const int m  = row0 + mo + r;
    const int bb = m >> 11;
    const int s  = m & (kS - 1);
    O[(((size_t)bb * kH + h) * kS + s) * kDh + d] = (__bf16)(acc[r] + bn);
  }
}

// ---------------------------------------------------------------------------
// Kernel 2: flash attention. Block = one (b,h) x 128 q rows (8 waves, one
// 16-row q tile each). K/V tiles (32 keys) staged in double-buffered LDS:
// K via GLOBAL_LOAD_ASYNC_TO_LDS_B128, V transposed manually.
// ---------------------------------------------------------------------------
__global__ void __launch_bounds__(256)
attn_kernel(const __bf16* __restrict__ Qb, const __bf16* __restrict__ Kb,
            const __bf16* __restrict__ Vb, __bf16* __restrict__ fused) {
  __shared__ __bf16 Kt[2][32 * 64];      // [key][d]   4KB each
  __shared__ __bf16 Vt[2][64 * 32];      // [d][key]   4KB each
  __shared__ __bf16 Pst[8][16 * 32];     // per-wave P staging
  const int tid  = threadIdx.x;
  const int lane = tid & 31;
  const int wisl = tid >> 5;

  constexpr int NQG = kS / 128;          // 16 q-groups per (b,h)
  const int bh = blockIdx.x / NQG;
  const int qg = blockIdx.x % NQG;
  const int hh = bh & (kH - 1);
  const int bb = bh >> 3;
  const __bf16* Qh = Qb + (size_t)bh * kS * kDh;
  const __bf16* Kh = Kb + (size_t)bh * kS * kDh;
  const __bf16* Vh = Vb + (size_t)bh * kS * kDh;
  const int q0 = qg * 128 + wisl * 16;

  // Q A-fragments (reused for every key tile)
  const v16bf qa0 = load_a_bf16v(Qh, kDh, q0, 0,  lane);
  const v16bf qa1 = load_a_bf16v(Qh, kDh, q0, 32, lane);

  // staging roles: each thread owns 8 contiguous d-elements of one key row
  const int frow = tid >> 3;             // 0..31 (key within tile)
  const int fseg = (tid & 7) * 8;        // d offset 0..56

  auto stage = [&](int buf, int key0) {
    // K: straight async copy, 16B per lane
    async_g2l_b128((unsigned)(size_t)&Kt[buf][frow * 64 + fseg],
                   Kh + (size_t)(key0 + frow) * kDh + fseg);
    // prefetch next V tile while we transpose this one
    __builtin_prefetch(Vh + (size_t)(key0 + 32 + frow) * kDh + fseg, 0, 1);
    // V: load b128, scatter-transpose into Vt[d][key]
    v8bf vv = *(const v8bf*)(Vh + (size_t)(key0 + frow) * kDh + fseg);
#pragma unroll
    for (int j = 0; j < 8; ++j) Vt[buf][(fseg + j) * 32 + frow] = vv[j];
  };

  v8f o0 = {}, o1 = {}, o2 = {}, o3 = {};
  float m8[8], l8[8];
#pragma unroll
  for (int r = 0; r < 8; ++r) { m8[r] = -3.0e38f; l8[r] = 0.0f; }

  __bf16* myp = Pst[wisl];
  const int n  = lane & 15;
  const int mo = (lane >> 4) << 3;
  const float sc = 0.125f;               // 1/sqrt(64)

  stage(0, 0);
  wait_asynccnt0();
  __syncthreads();

  constexpr int NKT = kS / 32;           // 64 key tiles
  for (int kc = 0; kc < NKT; ++kc) {
    const int cur = kc & 1;
    if (kc + 1 < NKT) stage(1 - cur, (kc + 1) * 32);
    const __bf16* Kc = Kt[cur];
    const __bf16* Vc = Vt[cur];

    // scores: two 16x16 C tiles over 32 keys; B-frag (k,c) = K[key][d]
    auto kfrag = [&](int keyrel, int kq0) {
      const __bf16* p = Kc + (keyrel + (lane & 15)) * 64 + kq0 + ((lane >> 4) << 3);
      return cat8(*(const v8bf*)p, *(const v8bf*)(p + 16));
    };
    v8f s0 = {}, s1 = {};
    s0 = wmma_bf16(qa0, kfrag(0, 0),   s0);
    s0 = wmma_bf16(qa1, kfrag(0, 32),  s0);
    s1 = wmma_bf16(qa0, kfrag(16, 0),  s1);
    s1 = wmma_bf16(qa1, kfrag(16, 32), s1);

    // online softmax (row = mo + r; stats replicated across 16-lane halves)
#pragma unroll
    for (int r = 0; r < 8; ++r) {
      float v0 = s0[r] * sc, v1 = s1[r] * sc;
      float mx = fmaxf(v0, v1);
#pragma unroll
      for (int dd = 1; dd < 16; dd <<= 1) mx = fmaxf(mx, __shfl_xor(mx, dd, 32));
      const float mnew = fmaxf(m8[r], mx);
      const float corr = __expf(m8[r] - mnew);
      const float p0 = __expf(v0 - mnew);
      const float p1 = __expf(v1 - mnew);
      float ps = p0 + p1;
#pragma unroll
      for (int dd = 1; dd < 16; dd <<= 1) ps += __shfl_xor(ps, dd, 32);
      l8[r] = l8[r] * corr + ps;
      m8[r] = mnew;
      o0[r] *= corr; o1[r] *= corr; o2[r] *= corr; o3[r] *= corr;
      myp[(mo + r) * 32 + n]      = (__bf16)p0;
      myp[(mo + r) * 32 + n + 16] = (__bf16)p1;
    }

    // P as A-fragment (two ds b128 loads)
    v16bf pa;
    {
      const int rr  = lane & 15;
      const int kb2 = (lane >> 4) << 3;
      pa = cat8(*(const v8bf*)(myp + rr * 32 + kb2),
                *(const v8bf*)(myp + rr * 32 + 16 + kb2));
    }

    // O += P[16x32] * V[32x64]; V B-frag from transposed LDS tile
    auto vfrag = [&](int n0) {
      const __bf16* p = Vc + (n0 + (lane & 15)) * 32 + ((lane >> 4) << 3);
      return cat8(*(const v8bf*)p, *(const v8bf*)(p + 16));
    };
    o0 = wmma_bf16(pa, vfrag(0),  o0);
    o1 = wmma_bf16(pa, vfrag(16), o1);
    o2 = wmma_bf16(pa, vfrag(32), o2);
    o3 = wmma_bf16(pa, vfrag(48), o3);

    wait_asynccnt0();    // next K tile landed
    __syncthreads();     // all waves done reading cur, staging next complete
  }

  // normalize, store to fused [B, S, HID] (bf16)
#pragma unroll
  for (int r = 0; r < 8; ++r) {
    const float inv = 1.0f / l8[r];
    const int s = q0 + mo + r;
    __bf16* dst = fused + ((size_t)bb * kS + s) * kHid + hh * kDh;
    dst[n]      = (__bf16)(o0[r] * inv);
    dst[16 + n] = (__bf16)(o1[r] * inv);
    dst[32 + n] = (__bf16)(o2[r] * inv);
    dst[48 + n] = (__bf16)(o3[r] * inv);
  }
}

// ---------------------------------------------------------------------------
// Kernel 3: mid = fused @ W1 + b1  ([8192,512]x[512,64] -> bf16)
// ---------------------------------------------------------------------------
__global__ void __launch_bounds__(256)
out1_kernel(const __bf16* __restrict__ fused, const float* __restrict__ W1,
            const float* __restrict__ b1, __bf16* __restrict__ mid) {
  __shared__ __bf16 Bt[2][16 * 32];
  const int tid = threadIdx.x, lane = tid & 31, wisl = tid >> 5;
  constexpr int NT = kDh / 16;              // 4
  const int row0 = (blockIdx.x / NT) * 128 + wisl * 16;
  const int col0 = (blockIdx.x % NT) * 16;

  stage_b_tile(W1, kDh, 0, col0, Bt[0], tid);
  __syncthreads();

  v8f acc = {};
  constexpr int NK = kHid / 32;             // 16
  for (int kc = 0; kc < NK; ++kc) {
    if (kc + 1 < NK) stage_b_tile(W1, kDh, (kc + 1) * 32, col0, Bt[(kc + 1) & 1], tid);
    v16bf a = load_a_bf16v(fused, kHid, row0, kc * 32, lane);
    v16bf b = read_b_frag(Bt[kc & 1], lane);
    acc = wmma_bf16(a, b, acc);
    __syncthreads();
  }

  const int n  = col0 + (lane & 15);
  const int mo = (lane >> 4) << 3;
  const float bn = b1[n];
#pragma unroll
  for (int r = 0; r < 8; ++r)
    mid[(size_t)(row0 + mo + r) * kDh + n] = (__bf16)(acc[r] + bn);
}

// ---------------------------------------------------------------------------
// Kernel 4: out = mid @ W2 + b2  ([8192,64]x[64,2048] -> fp32 d_out)
// ---------------------------------------------------------------------------
__global__ void __launch_bounds__(256)
out2_kernel(const __bf16* __restrict__ mid, const float* __restrict__ W2,
            const float* __restrict__ b2, float* __restrict__ out) {
  __shared__ __bf16 Bt[2][16 * 32];
  const int tid = threadIdx.x, lane = tid & 31, wisl = tid >> 5;
  constexpr int NT = kOut2 / 16;            // 128
  const int row0 = (blockIdx.x / NT) * 128 + wisl * 16;
  const int col0 = (blockIdx.x % NT) * 16;

  stage_b_tile(W2, kOut2, 0, col0, Bt[0], tid);
  __syncthreads();

  v8f acc = {};
  constexpr int NK = kDh / 32;              // 2
  for (int kc = 0; kc < NK; ++kc) {
    if (kc + 1 < NK) stage_b_tile(W2, kOut2, (kc + 1) * 32, col0, Bt[(kc + 1) & 1], tid);
    v16bf a = load_a_bf16v(mid, kDh, row0, kc * 32, lane);
    v16bf b = read_b_frag(Bt[kc & 1], lane);
    acc = wmma_bf16(a, b, acc);
    __syncthreads();
  }

  const int n  = col0 + (lane & 15);
  const int mo = (lane >> 4) << 3;
  const float bn = b2[n];
#pragma unroll
  for (int r = 0; r < 8; ++r)
    out[(size_t)(row0 + mo + r) * kOut2 + n] = acc[r] + bn;
}

// ---------------------------------------------------------------------------
extern "C" void kernel_launch(void* const* d_in, const int* in_sizes, int n_in,
                              void* d_out, int out_size, void* d_ws, size_t ws_size,
                              hipStream_t stream) {
  const float* in1 = (const float*)d_in[0];
  const float* in2 = (const float*)d_in[1];
  const float* Wq  = (const float*)d_in[2];
  const float* bq  = (const float*)d_in[3];
  const float* Wk  = (const float*)d_in[4];
  const float* bk  = (const float*)d_in[5];
  const float* Wv  = (const float*)d_in[6];
  const float* bv  = (const float*)d_in[7];
  const float* W1  = (const float*)d_in[8];
  const float* b1  = (const float*)d_in[9];
  const float* W2  = (const float*)d_in[10];
  const float* b2  = (const float*)d_in[11];

  char* ws = (char*)d_ws;
  const size_t qkv_bytes = (size_t)kB * kH * kS * kDh * 2;   // 8 MB each
  __bf16* Qb    = (__bf16*)ws;  ws += qkv_bytes;
  __bf16* Kb    = (__bf16*)ws;  ws += qkv_bytes;
  __bf16* Vb    = (__bf16*)ws;  ws += qkv_bytes;
  __bf16* fused = (__bf16*)ws;  ws += (size_t)kM * kHid * 2; // 8 MB
  __bf16* mid   = (__bf16*)ws;  ws += (size_t)kM * kDh * 2;  // 1 MB

  qkv_kernel<<<3 * (kM / 128) * (kHid / 16), 256, 0, stream>>>(
      in1, in2, Wq, bq, Wk, bk, Wv, bv, Qb, Kb, Vb);

  attn_kernel<<<kB * kH * (kS / 128), 256, 0, stream>>>(Qb, Kb, Vb, fused);

  out1_kernel<<<(kM / 128) * (kDh / 16), 256, 0, stream>>>(fused, W1, b1, mid);

  out2_kernel<<<(kM / 128) * (kOut2 / 16), 256, 0, stream>>>(mid, W2, b2,
                                                             (float*)d_out);
}
